// BinaryResNetBlock_1219770712416
// MI455X (gfx1250) — 8-bit matrix operand order repaired and checked by reading, compiled, not run on hardware
//
#include <hip/hip_runtime.h>
#include <stdint.h>

typedef __attribute__((ext_vector_type(8))) int v8i;

#define Nn 32
#define Cc 256
#define Hh 56
#define Ww 56
#define Mm (Nn*Hh*Ww)
#define TOT (25690112L)
#define KW_ELEMS (Cc*Cc*9)
#define BPACK_DW (16*9*4*32*8)
#define BN_EPS 1e-5f

__global__ void binarize_nchw_to_nhwc(const float* __restrict__ x,
                                      int8_t* __restrict__ a8) {
    long i = (long)blockIdx.x * 256 + threadIdx.x;
    if (i >= TOT) return;
    int c = (int)(i & 255);
    long m = i >> 8;
    int w = (int)(m % Ww); long t = m / Ww;
    int h = (int)(t % Hh); int n = (int)(t / Hh);
    float v = x[(((long)n * Cc + c) * Hh + h) * Ww + w];
    a8[i] = v > 0.f ? (int8_t)1 : (v < 0.f ? (int8_t)-1 : (int8_t)0);
}

__global__ void absmean_reduce(const float* __restrict__ Wt,
                               float* __restrict__ acc, int nel) {
    __shared__ float sdata[256];
    float s = 0.f;
    for (int i = blockIdx.x * 256 + threadIdx.x; i < nel; i += gridDim.x * 256)
        s += fabsf(Wt[i]);
    sdata[threadIdx.x] = s;
    __syncthreads();
    for (int o = 128; o > 0; o >>= 1) {
        if (threadIdx.x < o) sdata[threadIdx.x] += sdata[threadIdx.x + o];
        __syncthreads();
    }
    if (threadIdx.x == 0) atomicAdd(acc, sdata[0]);
}

__global__ void pack_weights(const float* __restrict__ Wt,
                             uint32_t* __restrict__ bpack) {
    int i = blockIdx.x * 256 + threadIdx.x;
    if (i >= BPACK_DW) return;
    int v    = i & 7;        int rest = i >> 3;
    int lane = rest & 31;    rest >>= 5;
    int ch   = rest & 3;     rest >>= 2;
    int t    = rest % 9;
    int wv   = rest / 9;
    int n    = wv * 16 + (lane & 15);
    int half = lane >> 4;
    int kh = t / 3, kw = t % 3;
    uint32_t d = 0;
    for (int j = 0; j < 4; ++j) {
        int kl = (v >> 2) * 32 + half * 16 + (v & 3) * 4 + j;
        int c  = ch * 64 + kl;
        float wval = Wt[(((long)n * Cc + c) * 3 + kh) * 3 + kw];
        int8_t sb = wval > 0.f ? 1 : (wval < 0.f ? -1 : 0);
        d |= ((uint32_t)(uint8_t)sb) << (8 * j);
    }
    bpack[i] = d;
}

__global__ void zero_stats(float* __restrict__ p, int n) {
    int i = blockIdx.x * 256 + threadIdx.x;
    if (i < n) p[i] = 0.f;
}

__global__ __launch_bounds__(256) void conv_bin_wmma(
    const int8_t*   __restrict__ a8,
    const uint32_t* __restrict__ bpack,
    const float*    __restrict__ scaleAcc,
    float*          __restrict__ y,
    float*          __restrict__ sum,
    float*          __restrict__ sumsq)
{
    __shared__ uint4 AsT[64 * 4];
    __shared__ int tapOff[9 * 64];

    unsigned long long* As8 = (unsigned long long*)AsT;

    const int tid   = threadIdx.x;
    const int lane  = tid & 31;
    const int wv    = tid >> 5;
    const int mbase = blockIdx.x * 64;
    const int half  = lane >> 4;
    const int mrow  = lane & 15;

    for (int idx = tid; idx < 9 * 64; idx += 256) {
        int t = idx >> 6, r = idx & 63;
        int m = mbase + r;
        int w = m % Ww; int tmp = m / Ww;
        int h = tmp % Hh; int n = tmp / Hh;
        int hh = h + (t / 3) - 1;
        int ww = w + (t % 3) - 1;
        tapOff[idx] = ((unsigned)hh < (unsigned)Hh && (unsigned)ww < (unsigned)Ww)
                    ? (((n * Hh + hh) * Ww + ww) << 8)
                    : -1;
    }
    __syncthreads();

    const float scale = *scaleAcc * (1.0f / (float)KW_ELEMS);

    v8i acc[2][4];
    for (int nt = 0; nt < 2; ++nt)
        for (int st = 0; st < 4; ++st)
            acc[nt][st] = (v8i){0, 0, 0, 0, 0, 0, 0, 0};

    const int srow  = tid >> 2;
    const int spart = tid & 3;

    uint4 sreg = {0u, 0u, 0u, 0u};
    {
        int so = tapOff[srow];
        if (so >= 0)
            sreg = *(const uint4*)(a8 + (long)so + spart * 16);
    }

    for (int k = 0; k < 36; ++k) {
        const int t  = k >> 2;
        const int ch = k & 3;

        __syncthreads();
        AsT[tid] = sreg;
        __syncthreads();

        if (k < 35) {
            const int k2 = k + 1;
            const int so = tapOff[(k2 >> 2) * 64 + srow];
            uint4 v = {0u, 0u, 0u, 0u};
            if (so >= 0)
                v = *(const uint4*)(a8 + (long)so + (k2 & 3) * 64 + spart * 16);
            sreg = v;
        }

        v8i b[2];
        for (int nt = 0; nt < 2; ++nt) {
            const uint32_t* bp = bpack +
                (((((wv * 2 + nt) * 9 + t) * 4 + ch) * 32 + lane) << 3);
            uint4 bl0 = *(const uint4*)(bp);
            uint4 bl1 = *(const uint4*)(bp + 4);
            b[nt][0] = (int)bl0.x; b[nt][1] = (int)bl0.y;
            b[nt][2] = (int)bl0.z; b[nt][3] = (int)bl0.w;
            b[nt][4] = (int)bl1.x; b[nt][5] = (int)bl1.y;
            b[nt][6] = (int)bl1.z; b[nt][7] = (int)bl1.w;
        }

#pragma unroll
        for (int st = 0; st < 4; ++st) {
            const int row = st * 16 + mrow;
            unsigned long long q0 = As8[row * 8 + half * 2 + 0];
            unsigned long long q1 = As8[row * 8 + half * 2 + 1];
            unsigned long long q2 = As8[row * 8 + half * 2 + 4];
            unsigned long long q3 = As8[row * 8 + half * 2 + 5];
            v8i a;
            a[0] = (int)(uint32_t)q0; a[1] = (int)(uint32_t)(q0 >> 32);
            a[2] = (int)(uint32_t)q1; a[3] = (int)(uint32_t)(q1 >> 32);
            a[4] = (int)(uint32_t)q2; a[5] = (int)(uint32_t)(q2 >> 32);
            a[6] = (int)(uint32_t)q3; a[7] = (int)(uint32_t)(q3 >> 32);

            acc[0][st] = __builtin_amdgcn_wmma_i32_16x16x64_iu8(
                true, a, true, b[0], acc[0][st], false, false);
            acc[1][st] = __builtin_amdgcn_wmma_i32_16x16x64_iu8(
                true, a, true, b[1], acc[1][st], false, false);
        }
    }

    for (int nt = 0; nt < 2; ++nt) {
        const int n = (wv * 2 + nt) * 16 + mrow;
        float s = 0.f, s2 = 0.f;
        for (int st = 0; st < 4; ++st) {
            for (int r = 0; r < 8; ++r) {
                float v = scale * (float)acc[nt][st][r];
                int m = mbase + st * 16 + half * 8 + r;
                y[(long)m * Cc + n] = v;
                s += v; s2 += v * v;
            }
        }
        atomicAdd(&sum[n], s);
        atomicAdd(&sumsq[n], s2);
    }
}

__global__ void bn_finalize(const float* __restrict__ sum,
                            const float* __restrict__ sumsq,
                            const float* __restrict__ gamma,
                            float* __restrict__ mu,
                            float* __restrict__ ivg) {
    int c = threadIdx.x;
    float m   = sum[c]   * (1.0f / (float)Mm);
    float var = sumsq[c] * (1.0f / (float)Mm) - m * m;
    mu[c]  = m;
    ivg[c] = rsqrtf(var + BN_EPS) * gamma[c];
}

__global__ void fuse_bn_add(const float* __restrict__ y,
                            const float* __restrict__ mu,
                            const float* __restrict__ ivg,
                            const float* __restrict__ beta,
                            const float* __restrict__ skip,
                            float* __restrict__ outNCHW,
                            int8_t* __restrict__ a8next) {
    long i = (long)blockIdx.x * 256 + threadIdx.x;
    if (i >= TOT) return;
    int c = (int)(i & 255);
    long m = i >> 8;
    int w = (int)(m % Ww); long t = m / Ww;
    int h = (int)(t % Hh); int n = (int)(t / Hh);
    long nchw = (((long)n * Cc + c) * Hh + h) * Ww + w;
    float v = (y[i] - mu[c]) * ivg[c] + beta[c] + skip[nchw];
    outNCHW[nchw] = v;
    if (a8next)
        a8next[i] = v > 0.f ? (int8_t)1 : (v < 0.f ? (int8_t)-1 : (int8_t)0);
}

extern "C" void kernel_launch(void* const* d_in, const int* in_sizes, int n_in,
                              void* d_out, int out_size, void* d_ws, size_t ws_size,
                              hipStream_t stream) {
    const float* x  = (const float*)d_in[0];
    const float* W1 = (const float*)d_in[1];
    const float* g1 = (const float*)d_in[2];
    const float* b1 = (const float*)d_in[3];
    const float* W2 = (const float*)d_in[4];
    const float* g2 = (const float*)d_in[5];
    const float* b2 = (const float*)d_in[6];
    float* out = (float*)d_out;

    char* ws = (char*)d_ws;
    size_t o = 0;
    int8_t* a8      = (int8_t*)(ws + o);  o += (size_t)TOT;
    float*  y       = (float*)(ws + o);   o += (size_t)TOT * 4;
    float*  inner   = (float*)(ws + o);   o += (size_t)TOT * 4;
    uint32_t* bpack = (uint32_t*)(ws + o); o += (size_t)BPACK_DW * 4;
    float*  stats   = (float*)(ws + o);
    float* sAcc = stats;
    float* sum  = stats + 1;
    float* ssq  = stats + 257;
    float* mu   = stats + 513;
    float* ivg  = stats + 769;

    const unsigned EB = (unsigned)(TOT / 256);
    const unsigned PB = (BPACK_DW + 255) / 256;
    const unsigned CB = Mm / 64;

    zero_stats<<<3, 256, 0, stream>>>(stats, 513);
    absmean_reduce<<<256, 256, 0, stream>>>(W1, sAcc, KW_ELEMS);
    pack_weights<<<PB, 256, 0, stream>>>(W1, bpack);
    binarize_nchw_to_nhwc<<<EB, 256, 0, stream>>>(x, a8);
    conv_bin_wmma<<<CB, 256, 0, stream>>>(a8, bpack, sAcc, y, sum, ssq);
    bn_finalize<<<1, 256, 0, stream>>>(sum, ssq, g1, mu, ivg);
    fuse_bn_add<<<EB, 256, 0, stream>>>(y, mu, ivg, b1, x, inner, a8);

    zero_stats<<<3, 256, 0, stream>>>(stats, 513);
    absmean_reduce<<<256, 256, 0, stream>>>(W2, sAcc, KW_ELEMS);
    pack_weights<<<PB, 256, 0, stream>>>(W2, bpack);
    conv_bin_wmma<<<CB, 256, 0, stream>>>(a8, bpack, sAcc, y, sum, ssq);
    bn_finalize<<<1, 256, 0, stream>>>(sum, ssq, g2, mu, ivg);
    fuse_bn_add<<<EB, 256, 0, stream>>>(y, mu, ivg, b2, inner, out, nullptr);
}
